// MultiHeadCrossWindowAttention_52037823758966
// MI455X (gfx1250) — matrix operand order repaired and checked by reading, not compiled, not run on hardware
//
#include <hip/hip_runtime.h>
#include <hip/hip_bf16.h>

typedef __attribute__((ext_vector_type(16))) _Float16 v16h;
typedef __attribute__((ext_vector_type(8)))  float    v8f;

#define NEGV (-1000000000.0f)

union H16 { v16h v; float4 f[2]; };
union HPK4 { _Float16 h[4]; uint2 u; };

// A-matrix fragment (16x32 f16, M x K) from LDS stored [row][K] (K contiguous).
// ISA 7.12.2: lanes 0-15 M=0-15 K={0-7,16-23}; lanes 16-31 M=0-15 K={8-15,24-31}.
__device__ inline v16h fragA(const _Float16* base, int stride, int m, int k0, int lane) {
  H16 u;
  int row = m + (lane & 15);
  int kb  = k0 + ((lane & 16) >> 1);          // 0 or 8
  u.f[0] = *(const float4*)(base + row * stride + kb);
  u.f[1] = *(const float4*)(base + row * stride + kb + 16);
  return u.v;
}

// B-matrix fragment (32x16 f16, K x N) from LDS stored [col][K] (K contiguous).
// lanes 0-15: N=0-15, K={0-7,16-23}; lanes 16-31: N=0-15, K={8-15,24-31}.
__device__ inline v16h fragB(const _Float16* base, int stride, int ncol, int k0, int lane) {
  H16 u;
  int col = ncol + (lane & 15);
  int kb  = k0 + ((lane & 16) >> 1);          // 0 or 8
  u.f[0] = *(const float4*)(base + col * stride + kb);
  u.f[1] = *(const float4*)(base + col * stride + kb + 16);
  return u.v;
}

__device__ inline v8f wmma_f16(v16h a, v16h b, v8f c) {
  return __builtin_amdgcn_wmma_f32_16x16x32_f16(false, a, false, b, (short)0, c, false, false);
}

// ---------------------------------------------------------------------------
// Kernel 1: 1x1 conv as GEMM  mem[b,o,p] = sum_c conv_w[o,c]*cat[b,c,p]
// block = 256 thr (8 waves); block tile 64(M)x64(N), K staged in 32-chunks.
// Wave grid 4(M)x2(N); each wave: one A-frag reused across two 16x16 tiles
// -> 16 WMMAs per K-step per block.  Global staging via b128 loads.
// Fused GroupNorm partial sums (each 16-row tile lies in exactly one group).
// ---------------------------------------------------------------------------
__global__ __launch_bounds__(256)
void k_conv(const float* __restrict__ x, const float* __restrict__ skip,
            const float* __restrict__ conv_w,
            float* __restrict__ ws_mem, float* __restrict__ stats) {
  const int t    = threadIdx.x;
  const int wv   = t >> 5;
  const int lane = t & 31;
  const int p0   = blockIdx.x * 64;      // pixel tile
  const int m0   = blockIdx.y * 64;      // out-channel tile
  const int bz   = blockIdx.z;

  __shared__ alignas(16) _Float16 Ablk[64 * 40];
  __shared__ alignas(16) _Float16 Bblk[64 * 40];

  const int tm = (wv >> 1) << 4;         // 0,16,32,48
  const int tn = (wv & 1) << 5;          // 0 or 32

  v8f acc0 = {0.f, 0.f, 0.f, 0.f, 0.f, 0.f, 0.f, 0.f};
  v8f acc1 = {0.f, 0.f, 0.f, 0.f, 0.f, 0.f, 0.f, 0.f};

  const float* xb = x    + (((size_t)(bz << 8)) << 14);
  const float* sb = skip + (((size_t)(bz << 8)) << 14);

  for (int kk = 0; kk < 16; kk++) {
    int k0 = kk << 5;
    // A: 64x32 f32 -> f16, b128 global loads (2 float4 per thread)
    for (int i = 0; i < 2; i++) {
      int f  = t + (i << 8);             // float4 index in [0,512)
      int ol = f >> 3, k4 = (f & 7) << 2;
      float4 v = *(const float4*)(conv_w + (m0 + ol) * 512 + k0 + k4);
      HPK4 pk;
      pk.h[0] = (_Float16)v.x; pk.h[1] = (_Float16)v.y;
      pk.h[2] = (_Float16)v.z; pk.h[3] = (_Float16)v.w;
      *(uint2*)(&Ablk[ol * 40 + k4]) = pk.u;
    }
    // B: 32(K) x 64(N), stored [n][k]; b128 loads along pixels
    for (int i = 0; i < 2; i++) {
      int f  = t + (i << 8);             // float4 index in [0,512)
      int n4 = (f & 15) << 2, kl = f >> 4;
      int ch = k0 + kl;
      const float* src = (ch < 256) ? (xb + (ch << 14))
                                    : (sb + ((ch - 256) << 14));
      float4 v = *(const float4*)(src + p0 + n4);
      Bblk[(n4 + 0) * 40 + kl] = (_Float16)v.x;
      Bblk[(n4 + 1) * 40 + kl] = (_Float16)v.y;
      Bblk[(n4 + 2) * 40 + kl] = (_Float16)v.z;
      Bblk[(n4 + 3) * 40 + kl] = (_Float16)v.w;
    }
    // prefetch next K-chunk of B into cache (global_prefetch_b8)
    if (kk < 15) {
      int ch = k0 + 32 + (t >> 4);
      const float* src = (ch < 256) ? (xb + (ch << 14))
                                    : (sb + ((ch - 256) << 14));
      __builtin_prefetch(src + p0 + ((t & 15) << 2), 0, 0);
    }
    __syncthreads();
    v16h a  = fragA(Ablk, 40, tm, 0, lane);
    acc0 = wmma_f16(a, fragB(Bblk, 40, tn,      0, lane), acc0);
    acc1 = wmma_f16(a, fragB(Bblk, 40, tn + 16, 0, lane), acc1);
    __syncthreads();
  }

  // write mem + GroupNorm partial sums
  float* memp = ws_mem + (((size_t)(bz << 8)) << 14);
  int pA = p0 + tn + (lane & 15);
  int pB = pA + 16;
  float s = 0.f, sq = 0.f;
  for (int r = 0; r < 8; r++) {
    int oo = m0 + tm + r + ((lane >> 4) << 3);
    float v0 = acc0[r], v1 = acc1[r];
    memp[(oo << 14) + pA] = v0;
    memp[(oo << 14) + pB] = v1;
    s += v0 + v1; sq += v0 * v0 + v1 * v1;
  }
  for (int off = 16; off > 0; off >>= 1) {
    s  += __shfl_xor(s,  off, 32);
    sq += __shfl_xor(sq, off, 32);
  }
  if (lane == 0) {
    int g = (m0 + tm) >> 4;              // group of this 16-row tile
    atomicAdd(&stats[bz * 16 + g], s);
    atomicAdd(&stats[128 + bz * 16 + g], sq);
  }
}

// ---------------------------------------------------------------------------
// Kernel 2: finalize GroupNorm stats: stats[i]=mean, stats[128+i]=rstd
// ---------------------------------------------------------------------------
__global__ void k_stats(float* __restrict__ stats) {
  int i = threadIdx.x;
  if (i < 128) {
    const float n = 262144.0f;           // 16 ch * 128 * 128
    float mean = stats[i] / n;
    float var  = stats[128 + i] / n - mean * mean;
    stats[i] = mean;
    stats[128 + i] = rsqrtf(var + 1e-5f);
  }
}

// ---------------------------------------------------------------------------
// Kernel 3: windowed attention.  One block (4 waves / 128 thr) per
// (batch, head, window).  Full f16 WMMA chain with f32 accumulate.
// ---------------------------------------------------------------------------
__global__ __launch_bounds__(128)
void k_attn(const float* __restrict__ x, const float* __restrict__ skip,
            const float* __restrict__ pe,
            const float* __restrict__ Wq, const float* __restrict__ Wk,
            const float* __restrict__ Wv, const float* __restrict__ Wlin,
            float* __restrict__ attn_out, float* __restrict__ ws_att) {
  const int n  = blockIdx.x;             // window 0..255 (wy*16+wx)
  const int hh = blockIdx.y;
  const int bz = blockIdx.z;
  const int wy = n >> 4, wx = n & 15;
  const int t = threadIdx.x;
  const int wv = t >> 5, lane = t & 31;
  const int c0 = hh << 5;

  __shared__ alignas(16) _Float16 Wqh[32 * 40], Wkh[32 * 40], Wvh[32 * 40], Wlh[32 * 40];
  __shared__ alignas(16) _Float16 xh[64 * 40], shm[64 * 40];
  __shared__ alignas(16) _Float16 qh[64 * 40], kh[64 * 40], oh[64 * 40];
  __shared__ alignas(16) _Float16 vT[32 * 72];
  __shared__ alignas(16) float    qkf[64 * 72];
  __shared__ alignas(16) _Float16 ah[64 * 72];

  // weights -> f16 LDS, row-major [d][e] (= [N][K] for B fragments)
  for (int i = 0; i < 8; i++) {
    int idx = t + (i << 7);
    int d = idx >> 5, e = idx & 31;
    Wqh[d * 40 + e] = (_Float16)Wq[idx];
    Wkh[d * 40 + e] = (_Float16)Wk[idx];
    Wvh[d * 40 + e] = (_Float16)Wv[idx];
    Wlh[d * 40 + e] = (_Float16)Wlin[idx];
  }
  // shifted-window gather (roll by -4,-4): [r][d] f16
  const size_t xbase = ((size_t)(bz * 256 + c0)) << 14;
  for (int i = 0; i < 16; i++) {
    int idx = t + (i << 7);
    int d = idx >> 6, r = idx & 63;
    int ri = r >> 3, rj = r & 7;
    int py = ((wy << 3) + ri + 4) & 127;
    int px = ((wx << 3) + rj + 4) & 127;
    int off = (d << 14) + (py << 7) + px;
    xh[r * 40 + d]  = (_Float16)x[xbase + off];
    shm[r * 40 + d] = (_Float16)skip[xbase + off];
  }
  __syncthreads();

  // q/k/v projections: 24 tiles of 16x16 (K=32), 6 per wave
  for (int tile = wv; tile < 24; tile += 4) {
    int mat = tile >> 3, sub = tile & 7;
    int tm = (sub >> 1) << 4;
    int tn = (sub & 1) << 4;
    const _Float16* A  = (mat == 0) ? xh : shm;
    const _Float16* Bm = (mat == 0) ? Wqh : ((mat == 1) ? Wkh : Wvh);
    v8f acc = {0.f, 0.f, 0.f, 0.f, 0.f, 0.f, 0.f, 0.f};
    acc = wmma_f16(fragA(A, 40, tm, 0, lane), fragB(Bm, 40, tn, 0, lane), acc);
    int col = tn + (lane & 15);
    for (int r = 0; r < 8; r++) {
      int row = tm + r + ((lane >> 4) << 3);
      _Float16 hv = (_Float16)acc[r];
      if (mat == 0)      qh[row * 40 + col] = hv;
      else if (mat == 1) kh[row * 40 + col] = hv;
      else               vT[col * 72 + row] = hv;   // transposed for attn@v
    }
  }
  __syncthreads();

  // qk^T / sqrt(hd) + pe bias + shift masks : 16 tiles, 4 per wave
  const float invs = 0.17677669529663687f;          // 1/sqrt(32)
  for (int tile = wv; tile < 16; tile += 4) {
    int tm = (tile >> 2) << 4, tn = (tile & 3) << 4;
    v8f acc = {0.f, 0.f, 0.f, 0.f, 0.f, 0.f, 0.f, 0.f};
    acc = wmma_f16(fragA(qh, 40, tm, 0, lane), fragB(kh, 40, tn, 0, lane), acc);
    int sc = tn + (lane & 15);
    int si = sc >> 3, sj = sc & 7;
    for (int r = 0; r < 8; r++) {
      int rr = tm + r + ((lane >> 4) << 3);
      int ri = rr >> 3, rj = rr & 7;
      float v = acc[r] * invs + pe[(si - ri + 7) * 64 + (sj - rj + 7)];
      if (wy == 15 && ((rr >= 32) != (sc >= 32))) v += NEGV;   // UL mask
      if (wx == 15 && ((rj >= 4)  != (sj >= 4)))  v += NEGV;   // LR mask
      qkf[rr * 72 + sc] = v;
    }
  }
  __syncthreads();

  // softmax rows (f32), write attn to global, f16 copy to LDS
  if (t < 64) {
    float mx = -3.4e38f;
    for (int s = 0; s < 64; s++) mx = fmaxf(mx, qkf[t * 72 + s]);
    float sum = 0.f;
    for (int s = 0; s < 64; s++) {
      float e = expf(qkf[t * 72 + s] - mx);
      qkf[t * 72 + s] = e; sum += e;
    }
    float inv = 1.0f / sum;
    int gbase = ((((bz << 3) + hh) << 8) + n) << 12;
    for (int s = 0; s < 64; s++) {
      float a = qkf[t * 72 + s] * inv;
      attn_out[gbase + (t << 6) + s] = a;
      ah[t * 72 + s] = (_Float16)a;
    }
  }
  __syncthreads();

  // attn @ v : 8 tiles (K=64), 2 per wave
  for (int tile = wv; tile < 8; tile += 4) {
    int tm = (tile >> 1) << 4, tn = (tile & 1) << 4;
    v8f acc = {0.f, 0.f, 0.f, 0.f, 0.f, 0.f, 0.f, 0.f};
    for (int ks = 0; ks < 64; ks += 32)
      acc = wmma_f16(fragA(ah, 72, tm, ks, lane), fragB(vT, 72, tn, ks, lane), acc);
    int col = tn + (lane & 15);
    for (int r = 0; r < 8; r++) {
      int row = tm + r + ((lane >> 4) << 3);
      oh[row * 40 + col] = (_Float16)acc[r];
    }
  }
  __syncthreads();

  // @ Wlin^T and scatter back with inverse roll (+4,+4)
  for (int tile = wv; tile < 8; tile += 4) {
    int tm = (tile >> 1) << 4, tn = (tile & 1) << 4;
    v8f acc = {0.f, 0.f, 0.f, 0.f, 0.f, 0.f, 0.f, 0.f};
    acc = wmma_f16(fragA(oh, 40, tm, 0, lane), fragB(Wlh, 40, tn, 0, lane), acc);
    int dp = tn + (lane & 15);
    float* dst = ws_att + (((size_t)(bz * 256 + c0 + dp)) << 14);
    for (int r = 0; r < 8; r++) {
      int row = tm + r + ((lane >> 4) << 3);
      int ri = row >> 3, rj = row & 7;
      int py = ((wy << 3) + ri + 4) & 127;
      int px = ((wx << 3) + rj + 4) & 127;
      dst[(py << 7) + px] = acc[r];
    }
  }
}

// ---------------------------------------------------------------------------
// Kernel 4: out = attn_branch + gelu(groupnorm(mem))   (float4 vectorized)
// ---------------------------------------------------------------------------
__global__ __launch_bounds__(256)
void k_final(const float* __restrict__ ws_mem, const float* __restrict__ ws_att,
             const float* __restrict__ stats,
             const float* __restrict__ gn_w, const float* __restrict__ gn_b,
             float* __restrict__ out) {
  int e = blockIdx.x * blockDim.x + threadIdx.x;   // float4 index, 8388608 total
  int ch = (e >> 12) & 255;
  int b  = e >> 20;
  int sgi = (b << 4) + (ch >> 4);
  float mean = stats[sgi], rstd = stats[128 + sgi];
  float w = gn_w[ch], bb = gn_b[ch];
  float4 mv = ((const float4*)ws_mem)[e];
  float4 av = ((const float4*)ws_att)[e];
  float4 o;
  {
    float v = (mv.x - mean) * rstd * w + bb;
    o.x = av.x + 0.5f * v * (1.0f + erff(v * 0.70710678118f));
    v = (mv.y - mean) * rstd * w + bb;
    o.y = av.y + 0.5f * v * (1.0f + erff(v * 0.70710678118f));
    v = (mv.z - mean) * rstd * w + bb;
    o.z = av.z + 0.5f * v * (1.0f + erff(v * 0.70710678118f));
    v = (mv.w - mean) * rstd * w + bb;
    o.w = av.w + 0.5f * v * (1.0f + erff(v * 0.70710678118f));
  }
  ((float4*)out)[e] = o;
}

// ---------------------------------------------------------------------------
extern "C" void kernel_launch(void* const* d_in, const int* in_sizes, int n_in,
                              void* d_out, int out_size, void* d_ws, size_t ws_size,
                              hipStream_t stream) {
  const float* skip   = (const float*)d_in[0];
  const float* x      = (const float*)d_in[1];
  const float* pe     = (const float*)d_in[2];
  const float* Wq     = (const float*)d_in[3];
  const float* Wk     = (const float*)d_in[4];
  const float* Wv     = (const float*)d_in[5];
  const float* Wlin   = (const float*)d_in[6];
  const float* conv_w = (const float*)d_in[7];
  const float* gn_w   = (const float*)d_in[8];
  const float* gn_b   = (const float*)d_in[9];

  float* out  = (float*)d_out;
  float* attn = out + 33554432;                    // 8*256*128*128

  float* ws_mem = (float*)d_ws;
  float* ws_att = ws_mem + 33554432;
  float* stats  = ws_att + 33554432;

  hipMemsetAsync(stats, 0, 256 * sizeof(float), stream);

  k_conv<<<dim3(256, 4, 8), 256, 0, stream>>>(x, skip, conv_w, ws_mem, stats);
  k_stats<<<1, 128, 0, stream>>>(stats);
  k_attn<<<dim3(256, 8, 8), 128, 0, stream>>>(x, skip, pe, Wq, Wk, Wv, Wlin, attn, ws_att);
  k_final<<<32768, 256, 0, stream>>>(ws_mem, ws_att, stats, gn_w, gn_b, out);
}
